// TransformerTower_23871428231617
// MI455X (gfx1250) — compile-verified, matrix operand order checked by reading, not run on hardware
//
#include <hip/hip_runtime.h>
#include <math.h>
#include <stdint.h>

#define B_    2
#define SEQ   2048
#define DIM_  1536
#define NH    8
#define DQK_  128
#define DV_   192
#define QKVN  1344
#define FFI   3072
#define DPW   128
#define PBR   128
#define EPSF  1.1920929e-07f
#define SCALEQ 0.125f

typedef unsigned short ushort_t;
typedef __attribute__((ext_vector_type(16))) __bf16 v16bf;
typedef __attribute__((ext_vector_type(8)))  float  v8f;
typedef __attribute__((ext_vector_type(4)))  unsigned int u32x4;

struct alignas(16) Frag32B { u32x4 a, b; };

static __device__ __forceinline__ unsigned short f2bf(float f) {
  unsigned int u = __float_as_uint(f);
  u += 0x7fffu + ((u >> 16) & 1u);
  return (unsigned short)(u >> 16);
}

static __device__ __forceinline__ v8f zero8() {
  v8f v;
  for (int i = 0; i < 8; ++i) v[i] = 0.0f;
  return v;
}

// CDNA5 async global->LDS copy, 16 bytes per lane, tracked by ASYNCcnt.
static __device__ __forceinline__ void asyncB128(const ushort_t* lds, const ushort_t* g) {
  unsigned ldsOff = (unsigned)(uintptr_t)lds;   // LDS aperture: addr[31:0] = LDS offset
  asm volatile("global_load_async_to_lds_b128 %0, %1, off"
               :: "v"(ldsOff), "v"(g) : "memory");
}
static __device__ __forceinline__ void waitAsync() {
  asm volatile("s_wait_asynccnt 0" ::: "memory");
}

// Load a 16x32 bf16 fragment from LDS laid out row-major [rows][ld].
// Matches CDNA5 16-bit A layout (and B layout when rows==N, i.e. B transposed):
// lane l: row = row0 + (l&15); elems e<8 -> k0 + (l>>4)*8 + e ; e>=8 -> +16.
static __device__ __forceinline__ v16bf ldsFrag(const ushort_t* base, int row0, int k0, int ld) {
  const int lane = threadIdx.x & 31;
  const ushort_t* p = base + (row0 + (lane & 15)) * ld + k0 + ((lane >> 4) << 3);
  Frag32B f;
  f.a = *(const u32x4*)p;
  f.b = *(const u32x4*)(p + 16);
  return __builtin_bit_cast(v16bf, f);
}

static __device__ __forceinline__ v8f wmma_bf16(v16bf a, v16bf b, v8f c) {
  return __builtin_amdgcn_wmma_f32_16x16x32_bf16(false, a, false, b, (short)0, c, false, false);
}

static __device__ __forceinline__ float ropeInv(int idx) {
  // inv_freq = 1/(idx + 10^linspace(1, 8129, 64))  (overflows to inf for idx>0, faithful)
  float lin = 1.0f + (float)idx * (8128.0f / 63.0f);
  return 1.0f / ((float)idx + powf(10.0f, lin));
}

// ---------------------------------------------------------------------------
// bf16 WMMA GEMM: C[M,N] = A[M,K] @ Bt[N,K]^T  (+bias, +add, relu)
// A row-major [M][K] bf16; Bt TRANSPOSED row-major [N][K] bf16.
// Block tile 128(M)x64(N), K-step 32, 8 waves, double-buffered async staging.
// ---------------------------------------------------------------------------
__global__ __launch_bounds__(256) void gemm_bf16(
    const ushort_t* __restrict__ A, const ushort_t* __restrict__ Bt,
    const float* __restrict__ bias, const float* __restrict__ add,
    float* __restrict__ outF, ushort_t* __restrict__ outH,
    int M, int N, int K, int relu)
{
  __shared__ ushort_t Al[2][128 * 32];  // [row][k], chunk c -> lin off c*8
  __shared__ ushort_t Bl[2][64 * 32];   // [n][k],  chunk c -> lin off c*8
  const int t = threadIdx.x;
  const int wid = t >> 5;
  const int lane = t & 31;
  const int m0 = blockIdx.y * 128;
  const int n0 = blockIdx.x * 64;
  const int k8 = (t & 3) << 3;          // k sub-offset for this thread's chunks

  const ushort_t* aSrc0 = A  + (size_t)(m0 + (t >> 2)) * K + k8;          // chunk t
  const ushort_t* aSrc1 = A  + (size_t)(m0 + ((t + 256) >> 2)) * K + k8;  // chunk t+256
  const ushort_t* bSrc  = Bt + (size_t)(n0 + (t >> 2)) * K + k8;          // chunk t

  auto issueTile = [&](int p, int k0) {
    asyncB128(&Al[p][t * 8],           aSrc0 + k0);
    asyncB128(&Al[p][(t + 256) * 8],   aSrc1 + k0);
    asyncB128(&Bl[p][t * 8],           bSrc  + k0);
  };

  v8f acc[4];
  for (int c = 0; c < 4; ++c) acc[c] = zero8();

  issueTile(0, 0);
  waitAsync();
  __syncthreads();

  const int nk = K >> 5;
  for (int ki = 0; ki < nk; ++ki) {
    const int p = ki & 1;
    if (ki + 1 < nk) issueTile(1 - p, (ki + 1) << 5);
    if (ki + 2 < nk) {
      __builtin_prefetch(aSrc0 + ((ki + 2) << 5), 0, 2);
      __builtin_prefetch(bSrc  + ((ki + 2) << 5), 0, 2);
    }
    // pre-load ALL fragments, then issue the WMMA chain without DS stalls
    v16bf a  = ldsFrag(Al[p], wid * 16, 0, 32);
    v16bf b0 = ldsFrag(Bl[p],  0, 0, 32);
    v16bf b1 = ldsFrag(Bl[p], 16, 0, 32);
    v16bf b2 = ldsFrag(Bl[p], 32, 0, 32);
    v16bf b3 = ldsFrag(Bl[p], 48, 0, 32);
    acc[0] = wmma_bf16(a, b0, acc[0]);
    acc[1] = wmma_bf16(a, b1, acc[1]);
    acc[2] = wmma_bf16(a, b2, acc[2]);
    acc[3] = wmma_bf16(a, b3, acc[3]);
    if (ki + 1 < nk) waitAsync();
    __syncthreads();
  }

  const int half = lane >> 4, nn = lane & 15;
  for (int c = 0; c < 4; ++c) {
    for (int r = 0; r < 8; ++r) {
      int m = m0 + wid * 16 + r + 8 * half;
      int n = n0 + c * 16 + nn;
      size_t idx = (size_t)m * N + n;
      float v = acc[c][r];
      if (bias) v += bias[n];
      if (add)  v += add[idx];
      if (relu) v = fmaxf(v, 0.0f);
      if (outF) outF[idx] = v;
      if (outH) outH[idx] = f2bf(v);
    }
  }
}

// ---------------------------------------------------------------------------
// Row RMSNorm (cols=1536) -> bf16
// ---------------------------------------------------------------------------
__global__ __launch_bounds__(256) void rmsnorm_rows(
    const float* __restrict__ x, const float* __restrict__ g,
    ushort_t* __restrict__ out, int cols)
{
  __shared__ float wred[8];
  __shared__ float stot;
  const int row = blockIdx.x, t = threadIdx.x;
  const float* xr = x + (size_t)row * cols;
  float v[6];
  float ss = 0.0f;
  for (int i = 0; i < 6; ++i) { v[i] = xr[t + 256 * i]; ss += v[i] * v[i]; }
  for (int m = 16; m >= 1; m >>= 1) ss += __shfl_xor(ss, m, 32);
  if ((t & 31) == 0) wred[t >> 5] = ss;
  __syncthreads();
  if (t == 0) { float s = 0.0f; for (int i = 0; i < 8; ++i) s += wred[i]; stot = s; }
  __syncthreads();
  float rms = rsqrtf(stot / (float)cols + EPSF);
  for (int i = 0; i < 6; ++i)
    out[(size_t)row * cols + t + 256 * i] = f2bf(v[i] * rms * g[t + 256 * i]);
}

// ---------------------------------------------------------------------------
// fp32 weight [K][N] -> TRANSPOSED bf16 [N][K]  (32x32 LDS tiles)
// grid = (N/32, K/32), block = 256
// ---------------------------------------------------------------------------
__global__ __launch_bounds__(256) void cvt_tr(
    const float* __restrict__ in, ushort_t* __restrict__ out, int K, int N)
{
  __shared__ ushort_t tl[32 * 40];
  const int nt = blockIdx.x * 32, kt = blockIdx.y * 32;
  const int t = threadIdx.x;
  for (int i = 0; i < 4; ++i) {
    int e = t + 256 * i;
    int r = e >> 5, c = e & 31;      // r: k-local, c: n-local
    tl[c * 40 + r] = f2bf(in[(size_t)(kt + r) * N + nt + c]);
  }
  __syncthreads();
  for (int i = 0; i < 4; ++i) {
    int e = t + 256 * i;
    int r = e >> 5, c = e & 31;      // r: n-local, c: k-local
    out[(size_t)(nt + r) * K + kt + c] = tl[r * 40 + c];
  }
}

// ---------------------------------------------------------------------------
// Post-QKV: per-token rmsnorm(q,k,v) + scale(q) + rope(q,k) -> bf16 buffers
// Block = 320 threads = 10 waves: waves 0..7 = q heads, 8 = k, 9 = v
// v written TRANSPOSED: vo[b][dv][seq]
// ---------------------------------------------------------------------------
__global__ __launch_bounds__(320) void qkv_post(
    const float* __restrict__ qkv,
    const float* __restrict__ qg, const float* __restrict__ kg, const float* __restrict__ vg,
    ushort_t* __restrict__ qo, ushort_t* __restrict__ ko, ushort_t* __restrict__ vo)
{
  const int tok = blockIdx.x;                 // 0..4095
  const int b = tok >> 11, npos = tok & 2047;
  const int wid = threadIdx.x >> 5, lane = threadIdx.x & 31;
  const float* row = qkv + (size_t)tok * QKVN;

  if (wid < 9) {
    const float* src = row + (wid < 8 ? wid * 128 : 1024);
    const float* gam = (wid < 8) ? qg : kg;
    float x[4];
    float ss = 0.0f;
    for (int i = 0; i < 4; ++i) { x[i] = src[lane + 32 * i]; ss += x[i] * x[i]; }
    for (int m = 16; m >= 1; m >>= 1) ss += __shfl_xor(ss, m, 32);
    float rms = rsqrtf(ss / 128.0f + EPSF);
    float y[4];
    for (int i = 0; i < 4; ++i) y[i] = x[i] * rms * gam[lane + 32 * i];
    if (wid < 8) for (int i = 0; i < 4; ++i) y[i] *= SCALEQ;
    // rope: lane holds dims {lane, lane+32, lane+64, lane+96}; (d, d+64) pair in-lane
    float f0 = (float)npos * ropeInv(lane);
    float f1 = (float)npos * ropeInv(lane + 32);
    float c0 = cosf(f0), s0 = sinf(f0), c1 = cosf(f1), s1 = sinf(f1);
    float o0 = y[0] * c0 - y[2] * s0;
    float o2 = y[2] * c0 + y[0] * s0;
    float o1 = y[1] * c1 - y[3] * s1;
    float o3 = y[3] * c1 + y[1] * s1;
    ushort_t* d = (wid < 8)
        ? qo + (((size_t)(b * NH + wid)) * SEQ + npos) * DQK_
        : ko + ((size_t)b * SEQ + npos) * DQK_;
    d[lane]      = f2bf(o0);
    d[lane + 32] = f2bf(o1);
    d[lane + 64] = f2bf(o2);
    d[lane + 96] = f2bf(o3);
  } else {
    const float* src = row + 1152;
    float x[6];
    float ss = 0.0f;
    for (int i = 0; i < 6; ++i) { x[i] = src[lane + 32 * i]; ss += x[i] * x[i]; }
    for (int m = 16; m >= 1; m >>= 1) ss += __shfl_xor(ss, m, 32);
    float rms = rsqrtf(ss / 192.0f + EPSF);
    for (int i = 0; i < 6; ++i) {
      int d = lane + 32 * i;
      vo[((size_t)b * DV_ + d) * SEQ + npos] = f2bf(x[i] * rms * vg[d]);
    }
  }
}

// ---------------------------------------------------------------------------
// Pairwise bias head: rmsnorm -> exact GELU -> Linear(128 -> 8 heads)
// One 128-thread block per (b,i,j)
// ---------------------------------------------------------------------------
__global__ __launch_bounds__(128) void pair_bias(
    const float* __restrict__ pw, const float* __restrict__ bg,
    const float* __restrict__ wb, float* __restrict__ pb)
{
  __shared__ float red[128];
  const int bid = blockIdx.x;     // linear (b,i,j)
  const int t = threadIdx.x;
  float x = pw[(size_t)bid * DPW + t];
  red[t] = x * x;
  __syncthreads();
  for (int s = 64; s >= 1; s >>= 1) { if (t < s) red[t] += red[t + s]; __syncthreads(); }
  float ms = red[0] / 128.0f;
  __syncthreads();
  float xn = x * rsqrtf(ms + EPSF) * bg[t];
  float ge = 0.5f * xn * (1.0f + erff(xn * 0.70710678118654752f));
  for (int h = 0; h < NH; ++h) {
    red[t] = ge * wb[t * NH + h];
    __syncthreads();
    for (int s = 64; s >= 1; s >>= 1) { if (t < s) red[t] += red[t + s]; __syncthreads(); }
    if (t == 0) pb[(size_t)bid * NH + h] = red[0];
    __syncthreads();
  }
}

// ---------------------------------------------------------------------------
// Flash attention: block = (64 queries) x (head g) x (batch b), j-blocks of 32
// Double-buffered async K/V staging; sim 4x WMMA (K=128); PV 6x WMMA (K=32)
// vt is pre-transposed: [b][dv][seq]
// ---------------------------------------------------------------------------
__global__ __launch_bounds__(256) void attn_fwd(
    const ushort_t* __restrict__ q, const ushort_t* __restrict__ k,
    const ushort_t* __restrict__ vt, const float* __restrict__ pb,
    ushort_t* __restrict__ out)
{
  __shared__ ushort_t Ql[64 * 128];
  __shared__ ushort_t Kl[2][32 * 128];
  __shared__ ushort_t Vt[2][192 * 32];   // [dv][j]
  __shared__ float    Sl[64 * 32];
  __shared__ ushort_t Pl[64 * 32];
  __shared__ float alpha_s[64];
  __shared__ float l_s[64];

  const int t = threadIdx.x, wid = t >> 5, lane = t & 31;
  const int i0 = blockIdx.x * 64;
  const int g  = blockIdx.y;
  const int b  = blockIdx.z;
  const int mi = wid >> 1;     // query row-fragment 0..3
  const int cg = wid & 1;      // column group

  const ushort_t* kBase = k + (size_t)b * SEQ * DQK_;                 // [j][d]
  const ushort_t* vBase = vt + ((size_t)b * DV_ + (t >> 2)) * SEQ + ((t & 3) << 3);

  auto issueKV = [&](int p, int j0) {
    // K tile 32x128: 512 flat chunks, 2 per thread
    asyncB128(&Kl[p][t * 8],         kBase + (size_t)j0 * DQK_ + t * 8);
    asyncB128(&Kl[p][(t + 256) * 8], kBase + (size_t)j0 * DQK_ + (t + 256) * 8);
    // Vt tile 192x32: 768 chunks (dv = c>>2, j8 = (c&3)*8), 3 per thread
    asyncB128(&Vt[p][t * 8],         vBase + j0);
    asyncB128(&Vt[p][(t + 256) * 8], vBase + (size_t)64 * SEQ + j0);
    asyncB128(&Vt[p][(t + 512) * 8], vBase + (size_t)128 * SEQ + j0);
  };

  { // async-load Q tile 64x128 (1024 flat chunks, 4 per thread)
    const ushort_t* src = q + (((size_t)(b * NH + g)) * SEQ + i0) * DQK_;
    for (int i = 0; i < 4; ++i)
      asyncB128(&Ql[(t + 256 * i) * 8], src + (t + 256 * i) * 8);
  }
  issueKV(0, 0);
  waitAsync();
  __syncthreads();

  v8f acc[6];
  for (int c = 0; c < 6; ++c) acc[c] = zero8();
  float m_row = -1e30f, l_row = 0.0f;

  for (int ki = 0; ki < SEQ / 32; ++ki) {
    const int p = ki & 1;
    const int j0 = ki * 32;
    if (ki + 1 < SEQ / 32) issueKV(1 - p, j0 + 32);

    // sim fragment (16x16) per wave: rows mi*16, cols cg*16, K=128
    // pre-load all Q/K fragments, then the 4-WMMA accumulate chain
    v16bf a0 = ldsFrag(Ql, mi * 16,  0, 128);
    v16bf a1 = ldsFrag(Ql, mi * 16, 32, 128);
    v16bf a2 = ldsFrag(Ql, mi * 16, 64, 128);
    v16bf a3 = ldsFrag(Ql, mi * 16, 96, 128);
    v16bf k0 = ldsFrag(Kl[p], cg * 16,  0, 128);
    v16bf k1 = ldsFrag(Kl[p], cg * 16, 32, 128);
    v16bf k2 = ldsFrag(Kl[p], cg * 16, 64, 128);
    v16bf k3 = ldsFrag(Kl[p], cg * 16, 96, 128);
    v8f sf = zero8();
    sf = wmma_bf16(a0, k0, sf);
    sf = wmma_bf16(a1, k1, sf);
    sf = wmma_bf16(a2, k2, sf);
    sf = wmma_bf16(a3, k3, sf);
    {
      int half = lane >> 4, nn = lane & 15;
      for (int r = 0; r < 8; ++r)
        Sl[(mi * 16 + r + 8 * half) * 32 + cg * 16 + nn] = sf[r];
    }
    __syncthreads();
    // online softmax, one row per thread (t<64)
    if (t < 64) {
      int row = t;
      int ip = (i0 + row) >> 4;
      int jp = j0 >> 4;
      float b0 = pb[(((size_t)b * PBR + ip) * PBR + jp) * NH + g];
      float b1 = pb[(((size_t)b * PBR + ip) * PBR + jp + 1) * NH + g];
      float mx = -1e30f;
      for (int c2 = 0; c2 < 32; ++c2) {
        float s = Sl[row * 32 + c2] + (c2 < 16 ? b0 : b1);
        s = 5.0f * tanhf(s * 0.2f);
        Sl[row * 32 + c2] = s;
        mx = fmaxf(mx, s);
      }
      float m_new = fmaxf(m_row, mx);
      float alpha = __expf(m_row - m_new);
      float sum = 0.0f;
      for (int c2 = 0; c2 < 32; ++c2) {
        float pv = __expf(Sl[row * 32 + c2] - m_new);
        sum += pv;
        Pl[row * 32 + c2] = f2bf(pv);
      }
      l_row = l_row * alpha + sum;
      m_row = m_new;
      alpha_s[row] = alpha;
    }
    __syncthreads();
    // rescale O accumulators + PV (one K=32 WMMA per 16x16 output fragment)
    {
      int half = lane >> 4;
      float al[8];
      for (int r = 0; r < 8; ++r) al[r] = alpha_s[mi * 16 + r + 8 * half];
      for (int c = 0; c < 6; ++c)
        for (int r = 0; r < 8; ++r) acc[c][r] = acc[c][r] * al[r];
      // pre-load P and all V fragments, then 6 independent WMMAs
      v16bf pa = ldsFrag(Pl, mi * 16, 0, 32);
      v16bf v0 = ldsFrag(Vt[p], cg * 96 +  0, 0, 32);
      v16bf v1 = ldsFrag(Vt[p], cg * 96 + 16, 0, 32);
      v16bf v2 = ldsFrag(Vt[p], cg * 96 + 32, 0, 32);
      v16bf v3 = ldsFrag(Vt[p], cg * 96 + 48, 0, 32);
      v16bf v4 = ldsFrag(Vt[p], cg * 96 + 64, 0, 32);
      v16bf v5 = ldsFrag(Vt[p], cg * 96 + 80, 0, 32);
      acc[0] = wmma_bf16(pa, v0, acc[0]);
      acc[1] = wmma_bf16(pa, v1, acc[1]);
      acc[2] = wmma_bf16(pa, v2, acc[2]);
      acc[3] = wmma_bf16(pa, v3, acc[3]);
      acc[4] = wmma_bf16(pa, v4, acc[4]);
      acc[5] = wmma_bf16(pa, v5, acc[5]);
    }
    if (ki + 1 < SEQ / 32) waitAsync();
    __syncthreads();
  }

  if (t < 64) l_s[t] = l_row;
  __syncthreads();
  {
    int half = lane >> 4, nn = lane & 15;
    float li[8];
    for (int r = 0; r < 8; ++r) li[r] = 1.0f / l_s[mi * 16 + r + 8 * half];
    for (int c = 0; c < 6; ++c)
      for (int r = 0; r < 8; ++r) {
        int m = mi * 16 + r + 8 * half;
        int n = cg * 96 + c * 16 + nn;
        out[((size_t)b * SEQ + i0 + m) * DIM_ + g * DV_ + n] = f2bf(acc[c][r] * li[r]);
      }
  }
}

// ---------------------------------------------------------------------------
extern "C" void kernel_launch(void* const* d_in, const int* in_sizes, int n_in,
                              void* d_out, int out_size, void* d_ws, size_t ws_size,
                              hipStream_t stream)
{
  (void)in_sizes; (void)n_in; (void)out_size; (void)ws_size;
  const float* in_x   = (const float*)d_in[0];
  const float* in_pw  = (const float*)d_in[1];
  const float* w_qkv  = (const float*)d_in[2];
  const float* q_g    = (const float*)d_in[3];
  const float* k_g    = (const float*)d_in[4];
  const float* v_g    = (const float*)d_in[5];
  const float* b_g    = (const float*)d_in[6];
  const float* w_bias = (const float*)d_in[7];
  const float* w_out  = (const float*)d_in[8];
  const float* apg    = (const float*)d_in[9];
  const float* fpg    = (const float*)d_in[10];
  const float* w_ff1  = (const float*)d_in[11];
  const float* b_ff1  = (const float*)d_in[12];
  const float* w_ff2  = (const float*)d_in[13];
  const float* b_ff2  = (const float*)d_in[14];
  float* x = (float*)d_out;               // residual stream lives in d_out

  const int M = B_ * SEQ;                 // 4096
  char* ws = (char*)d_ws;
  size_t off = 0;
  auto alloc = [&](size_t bytes) {
    char* p = ws + off;
    off += (bytes + 255) & ~(size_t)255;
    return p;
  };
  ushort_t* xn  = (ushort_t*)alloc((size_t)M * DIM_ * 2);
  ushort_t* wb  = (ushort_t*)alloc((size_t)DIM_ * FFI * 2);   // reused weight buffer (transposed bf16)
  float*    qkv = (float*)   alloc((size_t)M * QKVN * 4);
  ushort_t* qb  = (ushort_t*)alloc((size_t)B_ * NH * SEQ * DQK_ * 2);
  ushort_t* kb  = (ushort_t*)alloc((size_t)B_ * SEQ * DQK_ * 2);
  ushort_t* vb  = (ushort_t*)alloc((size_t)B_ * SEQ * DV_ * 2);
  float*    pbb = (float*)   alloc((size_t)B_ * PBR * PBR * NH * 4);
  ushort_t* ao  = (ushort_t*)alloc((size_t)M * DIM_ * 2);
  ushort_t* hb  = (ushort_t*)alloc((size_t)M * FFI * 2);

  hipMemcpyAsync(x, in_x, (size_t)M * DIM_ * sizeof(float),
                 hipMemcpyDeviceToDevice, stream);

  for (int d = 0; d < 8; ++d) {
    // ---- attention block ----
    rmsnorm_rows<<<M, 256, 0, stream>>>(x, apg + d * DIM_, xn, DIM_);
    cvt_tr<<<dim3(QKVN / 32, DIM_ / 32), 256, 0, stream>>>(
        w_qkv + (size_t)d * DIM_ * QKVN, wb, DIM_, QKVN);
    gemm_bf16<<<dim3(QKVN / 64, M / 128), 256, 0, stream>>>(
        xn, wb, nullptr, nullptr, qkv, nullptr, M, QKVN, DIM_, 0);
    qkv_post<<<M, 320, 0, stream>>>(qkv, q_g + d * DQK_, k_g + d * DQK_,
                                    v_g + d * DV_, qb, kb, vb);
    pair_bias<<<B_ * PBR * PBR, 128, 0, stream>>>(
        in_pw, b_g + d * DPW, w_bias + d * DPW * NH, pbb);
    attn_fwd<<<dim3(SEQ / 64, NH, B_), 256, 0, stream>>>(qb, kb, vb, pbb, ao);
    cvt_tr<<<dim3(DIM_ / 32, DIM_ / 32), 256, 0, stream>>>(
        w_out + (size_t)d * DIM_ * DIM_, wb, DIM_, DIM_);
    gemm_bf16<<<dim3(DIM_ / 64, M / 128), 256, 0, stream>>>(
        ao, wb, nullptr, x, x, nullptr, M, DIM_, DIM_, 0);   // residual add

    // ---- feed-forward block ----
    rmsnorm_rows<<<M, 256, 0, stream>>>(x, fpg + d * DIM_, xn, DIM_);
    cvt_tr<<<dim3(FFI / 32, DIM_ / 32), 256, 0, stream>>>(
        w_ff1 + (size_t)d * DIM_ * FFI, wb, DIM_, FFI);
    gemm_bf16<<<dim3(FFI / 64, M / 128), 256, 0, stream>>>(
        xn, wb, b_ff1 + d * FFI, nullptr, nullptr, hb, M, FFI, DIM_, 1);  // +bias, relu
    cvt_tr<<<dim3(DIM_ / 32, FFI / 32), 256, 0, stream>>>(
        w_ff2 + (size_t)d * FFI * DIM_, wb, FFI, DIM_);
    gemm_bf16<<<dim3(DIM_ / 64, M / 128), 256, 0, stream>>>(
        hb, wb, b_ff2 + d * DIM_, x, x, nullptr, M, DIM_, FFI, 0);        // +bias, residual
  }
}
